// MessageBlock_18932215841339
// MI455X (gfx1250) — matrix operand order repaired and checked by reading, not compiled, not run on hardware
//
#include <hip/hip_runtime.h>

// ---------------------------------------------------------------------------
// PaiNN message block, fused, for MI455X (gfx1250, wave32, WMMA).
//   phi = swish(s_j[src] @ W1 + b1) @ W2 + b2          (E x 384)
//   w_s = rbf(dist) @ Wd + bd                          (E x 384)
//   inv = phi * w_s  -> channels c%3: 0 -> *v_j[src], 1 -> delta_s, 2 -> *unit
//   segment-sum over dst via global f32 atomics (outputs fit in 192MB L2).
// All GEMM work runs on v_wmma_f32_16x16x32_f16; one 16-edge tile per wave.
// split0/split2 contributions to delta_v are merged via __shfl before the
// atomics (halves the dominant atomic traffic).
// ---------------------------------------------------------------------------

#define FEAT     128
#define F3       384
#define NRBF     20
#define KRBF     32          // NRBF padded to WMMA K
#define N_NODES  25000
#define N_EDGES  400000
#define NTILES   (N_EDGES / 16)   // 25000, exact
#define WAVES    8
#define BLOCK    (WAVES * 32)
#define NBLOCKS  320

typedef _Float16 half4  __attribute__((ext_vector_type(4)));
typedef _Float16 half8  __attribute__((ext_vector_type(8)));
typedef _Float16 v16h   __attribute__((ext_vector_type(16)));
typedef float    v8f    __attribute__((ext_vector_type(8)));
typedef float    f32x4  __attribute__((ext_vector_type(4)));

#define CAT16(lo, hi) __builtin_shufflevector(lo, hi, 0,1,2,3,4,5,6,7,8,9,10,11,12,13,14,15)

__device__ __forceinline__ void lds_fence() {
  asm volatile("s_wait_dscnt 0" ::: "memory");
}

__global__ void __launch_bounds__(BLOCK)
zero_f32(float* __restrict__ p, int n) {
  for (int i = blockIdx.x * blockDim.x + threadIdx.x; i < n;
       i += gridDim.x * blockDim.x)
    p[i] = 0.0f;
}

__global__ void __launch_bounds__(BLOCK)
painn_msg(const float* __restrict__ s_j, const float* __restrict__ v_j,
          const float* __restrict__ r_ij, const long long* __restrict__ nbrs,
          const float* __restrict__ W1, const float* __restrict__ b1,
          const float* __restrict__ W2, const float* __restrict__ b2,
          const float* __restrict__ Wd, const float* __restrict__ bd,
          float* __restrict__ out_s, float* __restrict__ out_v) {
  // ---- LDS: shared weights (f16, transposed: row = output col, contig K) ----
  __shared__ _Float16 sh_w1t[FEAT * FEAT];     // 32 KB
  __shared__ _Float16 sh_w2t[F3 * FEAT];       // 96 KB
  __shared__ _Float16 sh_wdt[F3 * KRBF];       // 24 KB (K padded with zeros)
  __shared__ float    sh_b1[FEAT];
  __shared__ float    sh_b2[F3];
  __shared__ float    sh_bd[F3];
  // ---- LDS: per-wave scratch ----
  __shared__ _Float16 sh_a[WAVES][16 * FEAT];  // A tile (s, then swish(h1))
  __shared__ _Float16 sh_rbf[WAVES][16 * KRBF];
  __shared__ int      sh_src[WAVES][16];
  __shared__ int      sh_dst[WAVES][16];
  __shared__ float    sh_unit[WAVES][16][3];
  __shared__ float    sh_dist[WAVES][16];

  const int tid  = threadIdx.x;
  const int lane = tid & 31;
  const int wv   = tid >> 5;
  const bool hi  = lane >= 16;
  const int  ln  = lane & 15;

  // ---------------- cooperative weight staging (once per block) -------------
  for (int i = tid; i < FEAT * FEAT; i += BLOCK) {
    int n = i >> 7, k = i & 127;
    sh_w1t[i] = (_Float16)W1[k * FEAT + n];
  }
  for (int i = tid; i < F3 * FEAT; i += BLOCK) {
    int c = i >> 7, k = i & 127;
    sh_w2t[i] = (_Float16)W2[k * F3 + c];
  }
  for (int i = tid; i < F3 * KRBF; i += BLOCK) {
    int c = i >> 5, k = i & 31;
    sh_wdt[i] = (k < NRBF) ? (_Float16)Wd[k * F3 + c] : (_Float16)0.0f;
  }
  for (int i = tid; i < FEAT; i += BLOCK) sh_b1[i] = b1[i];
  for (int i = tid; i < F3; i += BLOCK) { sh_b2[i] = b2[i]; sh_bd[i] = bd[i]; }
  __syncthreads();

  const int gw  = blockIdx.x * WAVES + wv;
  const int gws = gridDim.x * WAVES;

  for (int tile = gw; tile < NTILES; tile += gws) {
    // ---- 1. edge metadata (lanes 0..15: one edge each) ----
    if (lane < 16) {
      int e = tile * 16 + lane;
      int dsti = (int)nbrs[2 * e + 0];
      int srci = (int)nbrs[2 * e + 1];
      float rx = r_ij[3 * e + 0], ry = r_ij[3 * e + 1], rz = r_ij[3 * e + 2];
      float d  = sqrtf(rx * rx + ry * ry + rz * rz + 3.0f * 1e-15f);
      float id = __builtin_amdgcn_rcpf(d);             // v_rcp_f32
      sh_src[wv][lane] = srci;
      sh_dst[wv][lane] = dsti;
      sh_unit[wv][lane][0] = rx * id;
      sh_unit[wv][lane][1] = ry * id;
      sh_unit[wv][lane][2] = rz * id;
      sh_dist[wv][lane] = d;
    }
    lds_fence();

    // ---- 2. gather s_j rows -> f16 A tile; build RBF tile ----
    for (int j = lane; j < 16 * 32; j += 32) {       // 16 rows x 32 chunks
      int m = j >> 5, ch = j & 31;
      const f32x4 q = *(const f32x4*)(s_j + (size_t)sh_src[wv][m] * FEAT + ch * 4);
      half4 h = { (_Float16)q[0], (_Float16)q[1], (_Float16)q[2], (_Float16)q[3] };
      *(half4*)(&sh_a[wv][m * FEAT + ch * 4]) = h;
    }
    for (int j = lane; j < 16 * KRBF; j += 32) {
      int m = j >> 5, i = j & 31;
      float d  = sh_dist[wv][m];
      float id = __builtin_amdgcn_rcpf(d);
      float v = 0.0f;
      if (i < NRBF)
        v = __sinf((float)(i + 1) * (3.14159265358979323846f / 5.0f) * d) * id;
      sh_rbf[wv][j] = (_Float16)v;
    }
    lds_fence();

    // ---- 3. A1 fragments (ISA 16-bit A layout), kept in registers ----
    v16h a1[4];
    {
      const _Float16* at = &sh_a[wv][0];
#pragma unroll
      for (int ks = 0; ks < 4; ++ks) {
        int k0 = ks * 32 + (hi ? 8 : 0);
        half8 lo = *(const half8*)(at + ln * FEAT + k0);
        half8 h8 = *(const half8*)(at + ln * FEAT + k0 + 16);
        a1[ks] = CAT16(lo, h8);
      }
    }

    // ---- 4. GEMM1 (16x128 @ 128x128) + bias + swish -> back to LDS f16 ----
    const int mbase = hi ? 8 : 0;
#pragma unroll
    for (int t1 = 0; t1 < 8; ++t1) {
      v8f acc = {};
#pragma unroll
      for (int ks = 0; ks < 4; ++ks) {
        const _Float16* bp = &sh_w1t[(t1 * 16 + ln) * FEAT + ks * 32 + (hi ? 8 : 0)];
        half8 b0 = *(const half8*)bp;
        half8 b1v = *(const half8*)(bp + 16);
        v16h bf = CAT16(b0, b1v);
        acc = __builtin_amdgcn_wmma_f32_16x16x32_f16(false, a1[ks], false, bf,
                                                     (short)0, acc, false, false);
      }
      int c = t1 * 16 + ln;
      float bb = sh_b1[c];
#pragma unroll
      for (int r = 0; r < 8; ++r) {
        float x  = acc[r] + bb;
        // swish via fast rcp: x * 1/(1+exp(-x))  (v_exp_f32 + v_rcp_f32)
        float sw = x * __builtin_amdgcn_rcpf(1.0f + __expf(-x));
        sh_a[wv][(mbase + r) * FEAT + c] = (_Float16)sw;
      }
    }
    lds_fence();

    // ---- 5. A2 + RBF fragments ----
    v16h a2[4];
    {
      const _Float16* at = &sh_a[wv][0];
#pragma unroll
      for (int ks = 0; ks < 4; ++ks) {
        int k0 = ks * 32 + (hi ? 8 : 0);
        half8 lo = *(const half8*)(at + ln * FEAT + k0);
        half8 h8 = *(const half8*)(at + ln * FEAT + k0 + 16);
        a2[ks] = CAT16(lo, h8);
      }
    }
    v16h arbf;
    {
      const _Float16* rt = &sh_rbf[wv][ln * KRBF + (hi ? 8 : 0)];
      half8 lo = *(const half8*)rt;
      half8 h8 = *(const half8*)(rt + 16);
      arbf = CAT16(lo, h8);
    }

    // ---- per-lane row metadata in registers ----
    int   dstS[8], dstV[8], srcV[8];
    float ux[8], uy[8], uz[8];
#pragma unroll
    for (int r = 0; r < 8; ++r) {
      int m = mbase + r;
      int s = sh_src[wv][m], d = sh_dst[wv][m];
      srcV[r] = s * F3;
      dstS[r] = d * FEAT;
      dstV[r] = d * F3;
      ux[r] = sh_unit[wv][m][0];
      uy[r] = sh_unit[wv][m][1];
      uz[r] = sh_unit[wv][m][2];
    }

    // ---- 6. GEMM2 (W2) + RBF GEMM (Wd) per N-tile, fused consume ----
    for (int t2 = 0; t2 < 24; ++t2) {
      v8f accP = {};
#pragma unroll
      for (int ks = 0; ks < 4; ++ks) {
        const _Float16* bp = &sh_w2t[(t2 * 16 + ln) * FEAT + ks * 32 + (hi ? 8 : 0)];
        half8 b0 = *(const half8*)bp;
        half8 b1v = *(const half8*)(bp + 16);
        v16h bf = CAT16(b0, b1v);
        accP = __builtin_amdgcn_wmma_f32_16x16x32_f16(false, a2[ks], false, bf,
                                                      (short)0, accP, false, false);
      }
      v8f accW = {};
      {
        const _Float16* bp = &sh_wdt[(t2 * 16 + ln) * KRBF + (hi ? 8 : 0)];
        half8 b0 = *(const half8*)bp;
        half8 b1v = *(const half8*)(bp + 16);
        v16h bf = CAT16(b0, b1v);
        accW = __builtin_amdgcn_wmma_f32_16x16x32_f16(false, arbf, false, bf,
                                                      (short)0, accW, false, false);
      }

      int c    = t2 * 16 + ln;          // output column; fixed per lane
      float c2 = sh_b2[c], cd = sh_bd[c];
      int  mode = c % 3;
      int  f    = c / 3;
      int  f3   = f * 3;

      // every lane finishes its value first (uniform code -> shfl is legal)
      float val[8], val2[8];
#pragma unroll
      for (int r = 0; r < 8; ++r)
        val[r] = (accP[r] + c2) * (accW[r] + cd);
#pragma unroll
      for (int r = 0; r < 8; ++r)
        val2[r] = __shfl(val[r], lane + 2);   // partner column c+2 (same row)

      if (mode == 1) {                  // delta_s channel
#pragma unroll
        for (int r = 0; r < 8; ++r)
          atomicAdd(out_s + dstS[r] + f, val[r]);
      } else if (mode == 0) {           // delta_v: split0 * v_j (+ split2 * u)
        if (ln <= 13) {                 // partner (c+2) lives in this tile
#pragma unroll
          for (int r = 0; r < 8; ++r) {
            const float* vp = v_j + srcV[r] + f3;
            atomicAdd(out_v + dstV[r] + f3 + 0, val[r] * vp[0] + val2[r] * ux[r]);
            atomicAdd(out_v + dstV[r] + f3 + 1, val[r] * vp[1] + val2[r] * uy[r]);
            atomicAdd(out_v + dstV[r] + f3 + 2, val[r] * vp[2] + val2[r] * uz[r]);
          }
        } else {                        // partner in next tile: own part only
#pragma unroll
          for (int r = 0; r < 8; ++r) {
            const float* vp = v_j + srcV[r] + f3;
            atomicAdd(out_v + dstV[r] + f3 + 0, val[r] * vp[0]);
            atomicAdd(out_v + dstV[r] + f3 + 1, val[r] * vp[1]);
            atomicAdd(out_v + dstV[r] + f3 + 2, val[r] * vp[2]);
          }
        }
      } else {                          // mode 2: consumed by lane ln-2 unless
        if (ln < 2) {                   // the consumer was in the previous tile
#pragma unroll
          for (int r = 0; r < 8; ++r) {
            atomicAdd(out_v + dstV[r] + f3 + 0, val[r] * ux[r]);
            atomicAdd(out_v + dstV[r] + f3 + 1, val[r] * uy[r]);
            atomicAdd(out_v + dstV[r] + f3 + 2, val[r] * uz[r]);
          }
        }
      }
    }
  }
}

extern "C" void kernel_launch(void* const* d_in, const int* in_sizes, int n_in,
                              void* d_out, int out_size, void* d_ws, size_t ws_size,
                              hipStream_t stream) {
  (void)in_sizes; (void)n_in; (void)d_ws; (void)ws_size;
  const float*     s_j  = (const float*)d_in[0];
  const float*     v_j  = (const float*)d_in[1];
  const float*     r_ij = (const float*)d_in[2];
  const long long* nbrs = (const long long*)d_in[3];
  const float*     W1   = (const float*)d_in[4];
  const float*     b1   = (const float*)d_in[5];
  const float*     W2   = (const float*)d_in[6];
  const float*     b2   = (const float*)d_in[7];
  const float*     Wd   = (const float*)d_in[8];
  const float*     bd   = (const float*)d_in[9];

  float* out_s = (float*)d_out;                       // 25000 x 128
  float* out_v = out_s + (size_t)N_NODES * FEAT;      // 25000 x 128 x 3

  zero_f32<<<1024, BLOCK, 0, stream>>>((float*)d_out, out_size);
  painn_msg<<<NBLOCKS, BLOCK, 0, stream>>>(s_j, v_j, r_ij, nbrs,
                                           W1, b1, W2, b2, Wd, bd,
                                           out_s, out_v);
}
